// FlashAttention_11501922419239
// MI455X (gfx1250) — compile-verified, matrix operand order checked by reading, not run on hardware
//
#include <hip/hip_runtime.h>

// ---------------------------------------------------------------------------
// Flash attention with fused bf16 QKV projection for gfx1250 (MI455X).
//   qkv = x @ W^T + b  (fp32 in, bf16 out to workspace)
//   out = softmax(q k^T) @ v   (bf16 WMMA, fp32 accumulate, online softmax)
// Data movement: K tiles staged global->LDS via GLOBAL_LOAD_ASYNC_TO_LDS_B128
// (ASYNCcnt), V tiles staged transposed, both double-buffered; all WMMA
// operands fetched as two b128 loads (K-pairs are contiguous per lane).
// ---------------------------------------------------------------------------

#define N_TOK 8192
#define DIM   128
#define ATT_WAVES 4   // 4 waves * 16 rows = 64 query rows per block

typedef __attribute__((ext_vector_type(16))) __bf16 v16bf;
typedef __attribute__((ext_vector_type(8)))  float  v8f;

#if defined(__gfx1250__) && __has_builtin(__builtin_amdgcn_global_load_async_to_lds_b128)
#define HAVE_ASYNC_LDS 1
#else
#define HAVE_ASYNC_LDS 0
#endif

// Pointer types matching the async-LDS builtin's expected parameters:
// 16-byte int vector in global (AS1) / LDS (AS3) address spaces.
typedef int async_vec_t __attribute__((__vector_size__(16)));
typedef __attribute__((address_space(1))) async_vec_t* global_vec_p;
typedef __attribute__((address_space(3))) async_vec_t* lds_vec_p;

union ABf16 {
  uint4    q[2];
  unsigned u[8];
  __bf16   h[16];
  v16bf    v;
};

__device__ __forceinline__ v8f wmma_bf16(v16bf a, v16bf b, v8f c) {
  return __builtin_amdgcn_wmma_f32_16x16x32_bf16(false, a, false, b,
                                                 (short)0, c, false, false);
}

// 16-bit WMMA A/B operand layout: lane group lo/hi (lane>=16) holds
//   VGPR 0-3 : K = hi*8 + 0..7      (contiguous -> one b128)
//   VGPR 4-7 : K = 16 + hi*8 + 0..7 (contiguous -> one b128)
// `base` points at the operand's K=0 element for this lane's row/col.
__device__ __forceinline__ void load_operand(ABf16& m, const __bf16* base, int hi) {
  m.q[0] = *(const uint4*)(base + hi * 8);
  m.q[1] = *(const uint4*)(base + 16 + hi * 8);
}

__device__ __forceinline__ void wait_async_lds() {
#if HAVE_ASYNC_LDS
#if __has_builtin(__builtin_amdgcn_s_wait_asynccnt)
  __builtin_amdgcn_s_wait_asynccnt(0);
#else
  asm volatile("s_wait_asynccnt 0" ::: "memory");
#endif
#endif
}

// ---------------------------------------------------------------------------
// Kernel 1: qkv = x @ W^T + b, emitted as bf16 q/k/v planes in workspace.
// One 16x16 output tile per wave; K=128 -> 4 x wmma_f32_16x16x32_bf16.
// ---------------------------------------------------------------------------
__global__ __launch_bounds__(256) void qkv_proj_kernel(
    const float* __restrict__ x, const float* __restrict__ W,
    const float* __restrict__ bias, __bf16* __restrict__ qb,
    __bf16* __restrict__ kb, __bf16* __restrict__ vb) {
  const int lane = threadIdx.x & 31;
  const int wave = threadIdx.x >> 5;
  const int hi   = lane >> 4;
  const int ln16 = lane & 15;

  const int wg = blockIdx.x * 8 + wave;
  const int r0 = (wg / 24) * 16;   // row tile in [0, 8192)
  const int c0 = (wg % 24) * 16;   // col tile in [0, 384)

  v8f acc = {};
#pragma unroll
  for (int kc = 0; kc < 4; ++kc) {
    ABf16 a, bm;
    const float* xa = &x[(r0 + ln16) * DIM + kc * 32 + hi * 8];
    const float* wb = &W[(c0 + ln16) * DIM + kc * 32 + hi * 8];  // B(k,n)=W[c0+n][k]
    float4 f0 = *(const float4*)xa;
    float4 f1 = *(const float4*)(xa + 4);
    float4 f2 = *(const float4*)(xa + 16);
    float4 f3 = *(const float4*)(xa + 20);
    a.h[0] = (__bf16)f0.x;  a.h[1] = (__bf16)f0.y;  a.h[2] = (__bf16)f0.z;  a.h[3] = (__bf16)f0.w;
    a.h[4] = (__bf16)f1.x;  a.h[5] = (__bf16)f1.y;  a.h[6] = (__bf16)f1.z;  a.h[7] = (__bf16)f1.w;
    a.h[8] = (__bf16)f2.x;  a.h[9] = (__bf16)f2.y;  a.h[10] = (__bf16)f2.z; a.h[11] = (__bf16)f2.w;
    a.h[12] = (__bf16)f3.x; a.h[13] = (__bf16)f3.y; a.h[14] = (__bf16)f3.z; a.h[15] = (__bf16)f3.w;
    float4 g0 = *(const float4*)wb;
    float4 g1 = *(const float4*)(wb + 4);
    float4 g2 = *(const float4*)(wb + 16);
    float4 g3 = *(const float4*)(wb + 20);
    bm.h[0] = (__bf16)g0.x;  bm.h[1] = (__bf16)g0.y;  bm.h[2] = (__bf16)g0.z;  bm.h[3] = (__bf16)g0.w;
    bm.h[4] = (__bf16)g1.x;  bm.h[5] = (__bf16)g1.y;  bm.h[6] = (__bf16)g1.z;  bm.h[7] = (__bf16)g1.w;
    bm.h[8] = (__bf16)g2.x;  bm.h[9] = (__bf16)g2.y;  bm.h[10] = (__bf16)g2.z; bm.h[11] = (__bf16)g2.w;
    bm.h[12] = (__bf16)g3.x; bm.h[13] = (__bf16)g3.y; bm.h[14] = (__bf16)g3.z; bm.h[15] = (__bf16)g3.w;
    acc = wmma_bf16(a.v, bm.v, acc);
  }

  const float bv = bias[c0 + ln16];           // C layout: col = lane%16
  __bf16* dst = (c0 < 128) ? qb : (c0 < 256) ? kb : vb;
  const int cin = (c0 & 127) + ln16;
#pragma unroll
  for (int r = 0; r < 8; ++r)                 // C layout: row = r + 8*hi
    dst[(r0 + r + 8 * hi) * DIM + cin] = (__bf16)(acc[r] + bv);
}

// ---------------------------------------------------------------------------
// Stage one 32-key tile: K row-major via async global->LDS copy, V transposed
// via registers. Tile is contiguous: kb[j0*DIM .. (j0+32)*DIM).
// ---------------------------------------------------------------------------
__device__ __forceinline__ void stage_tile(const __bf16* __restrict__ kb,
                                           const __bf16* __restrict__ vb,
                                           int j0, __bf16* sKbuf, __bf16* sVTbuf,
                                           int tid, int nthreads) {
  for (int i = tid; i < (32 * DIM) / 8; i += nthreads) {
    const int e = i * 8;
#if HAVE_ASYNC_LDS
    __builtin_amdgcn_global_load_async_to_lds_b128(
        (global_vec_p)(kb + j0 * DIM + e),
        (lds_vec_p)(sKbuf + e), 0, 0);
#else
    *(uint4*)&sKbuf[e] = *(const uint4*)&kb[j0 * DIM + e];
#endif
    union { uint4 q; __bf16 h[8]; } vv;
    vv.q = *(const uint4*)&vb[j0 * DIM + e];
    const int row = i >> 4;
    const int col = (i & 15) * 8;
#pragma unroll
    for (int t = 0; t < 8; ++t) sVTbuf[(col + t) * 32 + row] = vv.h[t];
  }
}

// ---------------------------------------------------------------------------
// Kernel 2: flash attention. Each wave owns 16 query rows; double-buffered
// 32-key tiles. Per tile: 8 WMMAs for S, online softmax (shfl reductions),
// P transposed through per-wave LDS, 8 WMMAs for O += P@V.
// ---------------------------------------------------------------------------
__global__ __launch_bounds__(ATT_WAVES * 32) void flash_attn_kernel(
    const __bf16* __restrict__ qb, const __bf16* __restrict__ kb,
    const __bf16* __restrict__ vb, float* __restrict__ out) {
  __shared__ __align__(16) __bf16 sK[2][32 * DIM];    // K tile, row-major
  __shared__ __align__(16) __bf16 sVT[2][DIM * 32];   // V tile, transposed [d][n]
  __shared__ __align__(16) __bf16 sP[ATT_WAVES][16 * 32];

  const int tid  = threadIdx.x;
  const int lane = tid & 31;
  const int wave = tid >> 5;
  const int hi   = lane >> 4;
  const int ln16 = lane & 15;
  const int qRow0 = blockIdx.x * (ATT_WAVES * 16) + wave * 16;

  // Preload this wave's 16x128 Q tile in WMMA A layout (2 b128 loads/chunk).
  ABf16 qa[4];
#pragma unroll
  for (int kc = 0; kc < 4; ++kc)
    load_operand(qa[kc], &qb[(qRow0 + ln16) * DIM + kc * 32], hi);

  v8f zero = {};
  v8f o[8];
#pragma unroll
  for (int d = 0; d < 8; ++d) o[d] = zero;
  float mrow[8], lrow[8];
#pragma unroll
  for (int r = 0; r < 8; ++r) { mrow[r] = -3.0e38f; lrow[r] = 0.0f; }

  // Prologue: stage tile 0 into buffer 0.
  stage_tile(kb, vb, 0, sK[0], sVT[0], tid, ATT_WAVES * 32);
  wait_async_lds();
  __syncthreads();

  const int NT = N_TOK / 32;
  for (int jt = 0; jt < NT; ++jt) {
    const int p = jt & 1;
    if (jt + 1 < NT) {  // stage next tile into the other buffer while computing
      stage_tile(kb, vb, (jt + 1) * 32, sK[p ^ 1], sVT[p ^ 1], tid, ATT_WAVES * 32);
      __builtin_prefetch(kb + (jt + 2) * 32 * DIM + tid * 32, 0, 0);
      __builtin_prefetch(vb + (jt + 2) * 32 * DIM + tid * 32, 0, 0);
    }

    // S = Q @ K^T : two 16-column subtiles, K=128 in 4 chunks.
    v8f s0 = zero, s1 = zero;
#pragma unroll
    for (int kc = 0; kc < 4; ++kc) {
      ABf16 b0, b1;
      load_operand(b0, &sK[p][ln16 * DIM + kc * 32], hi);        // keys j0..j0+15
      load_operand(b1, &sK[p][(16 + ln16) * DIM + kc * 32], hi); // keys +16..+31
      s0 = wmma_bf16(qa[kc].v, b0.v, s0);
      s1 = wmma_bf16(qa[kc].v, b1.v, s1);
    }

    // Online softmax. C layout: VGPR r covers row (r + 8*hi), cols across 16 lanes.
    float sc[8];
#pragma unroll
    for (int r = 0; r < 8; ++r) {
      float mx = fmaxf(s0[r], s1[r]);
      mx = fmaxf(mx, __shfl_xor(mx, 1, 32));
      mx = fmaxf(mx, __shfl_xor(mx, 2, 32));
      mx = fmaxf(mx, __shfl_xor(mx, 4, 32));
      mx = fmaxf(mx, __shfl_xor(mx, 8, 32));
      const float mn = fmaxf(mrow[r], mx);
      const float scale = __expf(mrow[r] - mn);
      mrow[r] = mn;
      const float p0 = __expf(s0[r] - mn);
      const float p1 = __expf(s1[r] - mn);
      s0[r] = p0; s1[r] = p1;
      float rs = p0 + p1;
      rs += __shfl_xor(rs, 1, 32);
      rs += __shfl_xor(rs, 2, 32);
      rs += __shfl_xor(rs, 4, 32);
      rs += __shfl_xor(rs, 8, 32);
      lrow[r] = lrow[r] * scale + rs;
      sc[r] = scale;
    }
#pragma unroll
    for (int d = 0; d < 8; ++d)
#pragma unroll
      for (int r = 0; r < 8; ++r) o[d][r] *= sc[r];

    // Transpose P from C layout to A layout via per-wave LDS scratch.
#pragma unroll
    for (int r = 0; r < 8; ++r) {
      const int row = r + 8 * hi;
      sP[wave][row * 32 + ln16]      = (__bf16)s0[r];
      sP[wave][row * 32 + 16 + ln16] = (__bf16)s1[r];
    }
    ABf16 pa;
    load_operand(pa, &sP[wave][ln16 * 32], hi);

    // O += P @ V  (V^T staging makes K contiguous along LDS rows -> b128)
#pragma unroll
    for (int d = 0; d < 8; ++d) {
      ABf16 bv;
      load_operand(bv, &sVT[p][(d * 16 + ln16) * 32], hi);
      o[d] = wmma_bf16(pa.v, bv.v, o[d]);
    }

    wait_async_lds();  // our async writes into buffer p^1 have landed
    __syncthreads();   // everyone done reading p / writing p^1
  }

  float inv[8];
#pragma unroll
  for (int r = 0; r < 8; ++r) inv[r] = 1.0f / lrow[r];
#pragma unroll
  for (int d = 0; d < 8; ++d)
#pragma unroll
    for (int r = 0; r < 8; ++r)
      out[(qRow0 + r + 8 * hi) * DIM + d * 16 + ln16] = o[d][r] * inv[r];
}

// ---------------------------------------------------------------------------
extern "C" void kernel_launch(void* const* d_in, const int* in_sizes, int n_in,
                              void* d_out, int out_size, void* d_ws, size_t ws_size,
                              hipStream_t stream) {
  const float* x = (const float*)d_in[0];   // [8192, 128]
  const float* W = (const float*)d_in[1];   // [384, 128]
  const float* b = (const float*)d_in[2];   // [384]
  float* out = (float*)d_out;               // [8192, 128]

  __bf16* qb = (__bf16*)d_ws;               // 3 bf16 planes of [8192,128] = 6 MB
  __bf16* kb = qb + (size_t)N_TOK * DIM;
  __bf16* vb = kb + (size_t)N_TOK * DIM;

  // 512 row-tiles * 24 col-tiles = 12288 wave-tiles, 8 waves/block.
  const int projBlocks = (N_TOK / 16) * (3 * DIM / 16) / 8;
  qkv_proj_kernel<<<projBlocks, 256, 0, stream>>>(x, W, b, qb, kb, vb);

  const int attnBlocks = N_TOK / (ATT_WAVES * 16);
  flash_attn_kernel<<<attnBlocks, ATT_WAVES * 32, 0, stream>>>(qb, kb, vb, out);
}
